// MixtureOfExperts_88356067213562
// MI455X (gfx1250) — compile-verified, matrix operand order checked by reading, not run on hardware
//
#include <hip/hip_runtime.h>
#include <stdint.h>

// ---------------- problem constants ----------------
static constexpr int B_TOK = 16384;
static constexpr int D_IN  = 1024;
static constexpr int D_OUT = 1024;
static constexpr int H_DIM = 2048;
static constexpr int E_NUM = 8;
static constexpr int TOPK  = 2;

// ---------------- types ----------------
typedef __bf16 bf16;
typedef __attribute__((ext_vector_type(16))) __bf16 v16bf;
typedef __attribute__((ext_vector_type(8)))  __bf16 v8bf;
typedef __attribute__((ext_vector_type(8)))  float  v8f;
typedef __attribute__((ext_vector_type(4)))  unsigned u32x4;
typedef __attribute__((ext_vector_type(4)))  int      i32x4;
typedef __attribute__((ext_vector_type(8)))  int      i32x8;

__device__ __forceinline__ bf16 f2bf(float f) {
    unsigned u = __builtin_bit_cast(unsigned, f);
    unsigned r = u + 0x7FFFu + ((u >> 16) & 1u);   // round-to-nearest-even
    unsigned short h = (unsigned short)(r >> 16);
    return __builtin_bit_cast(bf16, h);
}

// Fragment loader: A (16x32 MxK) / B (32x16 KxN, symmetric layout).
// Lane L<16 owns row/col (L&15) with K {0..7, 16..23}; L>=16 owns K {8..15, 24..31}.
__device__ __forceinline__ v16bf load_frag(const bf16* base, int row, int kbase, int lane) {
    const bf16* p = base + row * 64 + kbase + ((lane & 16) ? 8 : 0);
    union { v16bf v; v8bf h[2]; } u;
    u.h[0] = *(const v8bf*)(p);
    u.h[1] = *(const v8bf*)(p + 16);
    return u.v;
}

// ---- Tensor Data Mover: 2-D tile (128 rows x 64 bf16 cols) from row-major matrix to LDS ----
// D# per CDNA5 ISA: group0 = count|lds_addr|global_addr|type=2,
// group1 = data_size, tensor_dim0/1, tile_dim0=64, tile_dim1=128, dim0_stride.
__device__ __forceinline__ void tdm_load_tile(unsigned lds_off, const bf16* gtile,
                                              unsigned Ktot, unsigned Ntot) {
    const uint64_t ga = (uint64_t)(uintptr_t)gtile;
    u32x4 g0;
    g0[0] = 1u;                                                    // count=1 (valid user D#)
    g0[1] = lds_off;                                               // lds_addr (bytes)
    g0[2] = (unsigned)ga;                                          // global_addr[31:0]
    g0[3] = (unsigned)((ga >> 32) & 0x01FFFFFFu) | 0x80000000u;    // addr[56:32] | type=2
    i32x8 g1;
    g1[0] = 0x00010000;                                            // data_size=1 (2 bytes)
    g1[1] = (int)((Ktot & 0xFFFFu) << 16);                         // tensor_dim0[15:0]
    g1[2] = (int)((Ktot >> 16) | ((Ntot & 0xFFFFu) << 16));        // dim0 hi | tensor_dim1 lo
    g1[3] = (int)((Ntot >> 16) | (64u << 16));                     // dim1 hi | tile_dim0=64
    g1[4] = 128;                                                   // tile_dim1=128
    g1[5] = (int)Ktot;                                             // tensor_dim0_stride[31:0]
    g1[6] = 0;
    g1[7] = 0;
    i32x4 z = {0, 0, 0, 0};
#if __clang_major__ >= 23
    i32x8 z8 = {0, 0, 0, 0, 0, 0, 0, 0};
    __builtin_amdgcn_tensor_load_to_lds(g0, g1, z, z, z8, 0);
#else
    __builtin_amdgcn_tensor_load_to_lds(g0, g1, z, z, 0);
#endif
}

// ---------------- small kernels ----------------
__global__ void moe_zero_kernel(unsigned* cnt) {
    if (threadIdx.x < E_NUM) cnt[threadIdx.x] = 0u;
}

__global__ __launch_bounds__(256) void cvt_bf16_kernel(const float* __restrict__ in,
                                                       bf16* __restrict__ out, int n4) {
    const int i = blockIdx.x * 256 + threadIdx.x;
    if (i >= n4) return;
    const float4 v = ((const float4*)in)[i];
    union { bf16 b[4]; uint64_t u; } t;
    t.b[0] = f2bf(v.x); t.b[1] = f2bf(v.y); t.b[2] = f2bf(v.z); t.b[3] = f2bf(v.w);
    ((uint64_t*)out)[i] = t.u;
}

__global__ __launch_bounds__(256) void moe_router_kernel(
    const float* __restrict__ x, const float* __restrict__ rW, const float* __restrict__ rb,
    float* __restrict__ probs_out, unsigned* __restrict__ top_idx, float2* __restrict__ top_w,
    unsigned* __restrict__ cnt)
{
    __shared__ __align__(16) float sW[E_NUM][D_IN];   // 32 KB
    __shared__ float sb[E_NUM];
    const int tid = threadIdx.x;
    for (int i = tid; i < E_NUM * D_IN; i += 256) ((float*)sW)[i] = rW[i];
    if (tid < E_NUM) sb[tid] = rb[tid];
    __syncthreads();

    const int b = blockIdx.x * 256 + tid;
    const float4* xr = (const float4*)(x + (size_t)b * D_IN);
    float acc[E_NUM];
#pragma unroll
    for (int e = 0; e < E_NUM; ++e) acc[e] = 0.f;
    for (int d = 0; d < D_IN / 4; ++d) {
        const float4 xv = xr[d];
#pragma unroll
        for (int e = 0; e < E_NUM; ++e) {
            const float4 wv = *(const float4*)(&sW[e][d * 4]);
            acc[e] += xv.x * wv.x + xv.y * wv.y + xv.z * wv.z + xv.w * wv.w;
        }
    }
    float mx = -1e30f;
#pragma unroll
    for (int e = 0; e < E_NUM; ++e) { acc[e] += sb[e]; mx = fmaxf(mx, acc[e]); }
    float p[E_NUM], sum = 0.f;
#pragma unroll
    for (int e = 0; e < E_NUM; ++e) { p[e] = __expf(acc[e] - mx); sum += p[e]; }
    const float inv = 1.f / sum;
#pragma unroll
    for (int e = 0; e < E_NUM; ++e) probs_out[(size_t)b * E_NUM + e] = p[e] * inv;

    int e0 = 0; float p0 = p[0];
#pragma unroll
    for (int e = 1; e < E_NUM; ++e) if (p[e] > p0) { p0 = p[e]; e0 = e; }
    int e1 = (e0 == 0) ? 1 : 0; float p1 = p[e1];
#pragma unroll
    for (int e = 0; e < E_NUM; ++e) if (e != e0 && p[e] > p1) { p1 = p[e]; e1 = e; }
    const float rs = 1.f / (p0 + p1);
    top_idx[b] = (unsigned)e0 | ((unsigned)e1 << 8);
    top_w[b]   = make_float2(p0 * rs, p1 * rs);
    atomicAdd(&cnt[e0], 1u);
    atomicAdd(&cnt[e1], 1u);
}

__global__ void moe_prefix_kernel(const unsigned* __restrict__ cnt,
                                  unsigned* __restrict__ offs, unsigned* __restrict__ cursor) {
    if (threadIdx.x == 0) {
        unsigned s = 0;
        for (int e = 0; e < E_NUM; ++e) { offs[e] = s; cursor[e] = s; s += cnt[e]; }
    }
}

__global__ __launch_bounds__(256) void moe_scatter_kernel(
    const unsigned* __restrict__ top_idx, const float2* __restrict__ top_w,
    unsigned* __restrict__ cursor, int* __restrict__ tok_list, float* __restrict__ gate_list)
{
    const int b = blockIdx.x * 256 + threadIdx.x;
    const unsigned te = top_idx[b];
    const float2 w = top_w[b];
    unsigned q0 = atomicAdd(&cursor[te & 0xFFu], 1u);
    tok_list[q0] = b; gate_list[q0] = w.x;
    unsigned q1 = atomicAdd(&cursor[(te >> 8) & 0xFFu], 1u);
    tok_list[q1] = b; gate_list[q1] = w.y;
}

__global__ __launch_bounds__(256) void moe_biasinit_kernel(
    const float* __restrict__ b2, const unsigned* __restrict__ top_idx,
    const float2* __restrict__ top_w, float* __restrict__ out)
{
    const int b = blockIdx.x;
    const unsigned te = top_idx[b];
    const float2 w = top_w[b];
    const float* b2a = b2 + (size_t)(te & 0xFFu) * D_OUT;
    const float* b2b = b2 + (size_t)((te >> 8) & 0xFFu) * D_OUT;
    for (int o = threadIdx.x; o < D_OUT; o += 256)
        out[(size_t)b * D_OUT + o] = w.x * b2a[o] + w.y * b2b[o];
}

// ---------------- WMMA GEMMs (double-buffered, TDM for B tiles) ----------------
static constexpr int BM = 128, BN = 128, BK = 64;
static constexpr unsigned TILE_B = BM * BK * 2;       // 16 KB per buffer
static constexpr unsigned SB_OFF = 0;                 // sB[2] at 0, 16384
static constexpr unsigned SA_OFF = 2 * TILE_B;        // sA[2] at 32768, 49152
static constexpr unsigned SMEM_BYTES = 4 * TILE_B;    // 64 KB dynamic LDS (base offset 0)

// h[off_e + row, :] = relu( xbf[tok[row], :] @ W1bf[e]^T + b1[e] )   (stored bf16)
__global__ __launch_bounds__(256) void moe_gemm1_kernel(
    const bf16* __restrict__ xbf, const bf16* __restrict__ w1bf, const float* __restrict__ b1,
    const int* __restrict__ tok_list, const unsigned* __restrict__ cnt,
    const unsigned* __restrict__ offs, bf16* __restrict__ h_out)
{
    extern __shared__ __align__(16) char smem[];
    bf16* sB = (bf16*)(smem + SB_OFF);
    bf16* sA = (bf16*)(smem + SA_OFF);

    const int e = blockIdx.z;
    const int cnt_e = (int)cnt[e];
    const int m0 = blockIdx.x * BM;
    if (m0 >= cnt_e) return;
    const int off_e = (int)offs[e];
    const int n0 = blockIdx.y * BN;

    const int tid = threadIdx.x, lane = tid & 31, wid = tid >> 5;
    const int wm = (wid >> 2) * 64, wn = (wid & 3) * 32;

    // Per-thread A-staging: 4 chunks of 8 bf16 (16B). idx = tid + it*256; row = idx>>3, ch = idx&7.
    const bf16* ap[4];
#pragma unroll
    for (int it = 0; it < 4; ++it) {
        const int idx = tid + it * 256;
        int slot = m0 + (idx >> 3); if (slot >= cnt_e) slot = cnt_e - 1;
        ap[it] = xbf + (size_t)tok_list[off_e + slot] * D_IN + (idx & 7) * 8;
    }

    const bf16* w1e = w1bf + (size_t)e * H_DIM * D_IN + (size_t)n0 * D_IN;

    v8f czero = {};
    v8f c[4][2];
#pragma unroll
    for (int i = 0; i < 4; ++i)
#pragma unroll
        for (int j = 0; j < 2; ++j) c[i][j] = czero;

    constexpr int KT = D_IN / BK;   // 16 stages

    // prologue: stage 0
    if (wid == 0) tdm_load_tile(SB_OFF, w1e, D_IN, H_DIM);
    uint4 areg[4];
#pragma unroll
    for (int it = 0; it < 4; ++it) areg[it] = *(const uint4*)(ap[it]);
#pragma unroll
    for (int it = 0; it < 4; ++it) {
        const int idx = tid + it * 256;
        *(uint4*)(sA + (idx >> 3) * BK + (idx & 7) * 8) = areg[it];
    }
    if (wid == 0) __builtin_amdgcn_s_wait_tensorcnt(0);
    __syncthreads();

    for (int kt = 0; kt < KT; ++kt) {
        const int cur = kt & 1, nxt = cur ^ 1;
        const bool more = (kt + 1 < KT);
        if (more) {
            if (wid == 0) tdm_load_tile(SB_OFF + nxt * TILE_B, w1e + (kt + 1) * BK, D_IN, H_DIM);
#pragma unroll
            for (int it = 0; it < 4; ++it) areg[it] = *(const uint4*)(ap[it] + (kt + 1) * BK);
        }
        const bf16* A = sA + cur * (BM * BK);
        const bf16* Bm = sB + cur * (BN * BK);
#pragma unroll
        for (int ks = 0; ks < BK; ks += 32) {
            v16bf a[4], bfr[2];
#pragma unroll
            for (int i = 0; i < 4; ++i) a[i]   = load_frag(A,  wm + i * 16 + (lane & 15), ks, lane);
#pragma unroll
            for (int j = 0; j < 2; ++j) bfr[j] = load_frag(Bm, wn + j * 16 + (lane & 15), ks, lane);
#pragma unroll
            for (int i = 0; i < 4; ++i)
#pragma unroll
                for (int j = 0; j < 2; ++j)
                    c[i][j] = __builtin_amdgcn_wmma_f32_16x16x32_bf16(
                        false, a[i], false, bfr[j], (short)0, c[i][j], false, false);
        }
        if (more) {
#pragma unroll
            for (int it = 0; it < 4; ++it) {
                const int idx = tid + it * 256;
                *(uint4*)(sA + nxt * (BM * BK) + (idx >> 3) * BK + (idx & 7) * 8) = areg[it];
            }
            if (wid == 0) __builtin_amdgcn_s_wait_tensorcnt(0);
        }
        __syncthreads();
    }

    const int hi = (lane >> 4) << 3, col_l = lane & 15;
#pragma unroll
    for (int j = 0; j < 2; ++j) {
        const int n_g = n0 + wn + j * 16 + col_l;
        const float bias = b1[(size_t)e * H_DIM + n_g];
#pragma unroll
        for (int i = 0; i < 4; ++i) {
#pragma unroll
            for (int r = 0; r < 8; ++r) {
                const int m_l = wm + i * 16 + hi + r;
                if (m0 + m_l < cnt_e) {
                    float v = c[i][j][r] + bias;
                    v = v > 0.f ? v : 0.f;
                    h_out[(size_t)(off_e + m0 + m_l) * H_DIM + n_g] = f2bf(v);
                }
            }
        }
    }
}

// out[tok[row], :] += gate[row] * ( h[off_e + row, :] @ W2bf[e]^T )
__global__ __launch_bounds__(256) void moe_gemm2_kernel(
    const bf16* __restrict__ h_in, const bf16* __restrict__ w2bf,
    const int* __restrict__ tok_list, const float* __restrict__ gate_list,
    const unsigned* __restrict__ cnt, const unsigned* __restrict__ offs,
    float* __restrict__ out)
{
    extern __shared__ __align__(16) char smem[];
    bf16* sB = (bf16*)(smem + SB_OFF);
    bf16* sA = (bf16*)(smem + SA_OFF);

    const int e = blockIdx.z;
    const int cnt_e = (int)cnt[e];
    const int m0 = blockIdx.x * BM;
    if (m0 >= cnt_e) return;
    const int off_e = (int)offs[e];
    const int n0 = blockIdx.y * BN;

    const int tid = threadIdx.x, lane = tid & 31, wid = tid >> 5;
    const int wm = (wid >> 2) * 64, wn = (wid & 3) * 32;

    const bf16* ap[4];
#pragma unroll
    for (int it = 0; it < 4; ++it) {
        const int idx = tid + it * 256;
        int slot = m0 + (idx >> 3); if (slot >= cnt_e) slot = cnt_e - 1;
        ap[it] = h_in + (size_t)(off_e + slot) * H_DIM + (idx & 7) * 8;
    }

    const bf16* w2e = w2bf + (size_t)e * D_OUT * H_DIM + (size_t)n0 * H_DIM;

    v8f czero = {};
    v8f c[4][2];
#pragma unroll
    for (int i = 0; i < 4; ++i)
#pragma unroll
        for (int j = 0; j < 2; ++j) c[i][j] = czero;

    constexpr int KT = H_DIM / BK;   // 32 stages

    if (wid == 0) tdm_load_tile(SB_OFF, w2e, H_DIM, D_OUT);
    uint4 areg[4];
#pragma unroll
    for (int it = 0; it < 4; ++it) areg[it] = *(const uint4*)(ap[it]);
#pragma unroll
    for (int it = 0; it < 4; ++it) {
        const int idx = tid + it * 256;
        *(uint4*)(sA + (idx >> 3) * BK + (idx & 7) * 8) = areg[it];
    }
    if (wid == 0) __builtin_amdgcn_s_wait_tensorcnt(0);
    __syncthreads();

    for (int kt = 0; kt < KT; ++kt) {
        const int cur = kt & 1, nxt = cur ^ 1;
        const bool more = (kt + 1 < KT);
        if (more) {
            if (wid == 0) tdm_load_tile(SB_OFF + nxt * TILE_B, w2e + (kt + 1) * BK, H_DIM, D_OUT);
#pragma unroll
            for (int it = 0; it < 4; ++it) areg[it] = *(const uint4*)(ap[it] + (kt + 1) * BK);
        }
        const bf16* A = sA + cur * (BM * BK);
        const bf16* Bm = sB + cur * (BN * BK);
#pragma unroll
        for (int ks = 0; ks < BK; ks += 32) {
            v16bf a[4], bfr[2];
#pragma unroll
            for (int i = 0; i < 4; ++i) a[i]   = load_frag(A,  wm + i * 16 + (lane & 15), ks, lane);
#pragma unroll
            for (int j = 0; j < 2; ++j) bfr[j] = load_frag(Bm, wn + j * 16 + (lane & 15), ks, lane);
#pragma unroll
            for (int i = 0; i < 4; ++i)
#pragma unroll
                for (int j = 0; j < 2; ++j)
                    c[i][j] = __builtin_amdgcn_wmma_f32_16x16x32_bf16(
                        false, a[i], false, bfr[j], (short)0, c[i][j], false, false);
        }
        if (more) {
#pragma unroll
            for (int it = 0; it < 4; ++it) {
                const int idx = tid + it * 256;
                *(uint4*)(sA + nxt * (BM * BK) + (idx >> 3) * BK + (idx & 7) * 8) = areg[it];
            }
            if (wid == 0) __builtin_amdgcn_s_wait_tensorcnt(0);
        }
        __syncthreads();
    }

    const int hi = (lane >> 4) << 3, col_l = lane & 15;
#pragma unroll
    for (int i = 0; i < 4; ++i) {
#pragma unroll
        for (int r = 0; r < 8; ++r) {
            const int m_l = wm + i * 16 + hi + r;
            if (m0 + m_l < cnt_e) {
                const int   row  = off_e + m0 + m_l;
                const int   tokb = tok_list[row];
                const float g    = gate_list[row];
                float* orow = out + (size_t)tokb * D_OUT + n0 + wn;
#pragma unroll
                for (int j = 0; j < 2; ++j)
                    atomicAdd(orow + j * 16 + col_l, g * c[i][j][r]);
            }
        }
    }
}

// ---------------- launch ----------------
extern "C" void kernel_launch(void* const* d_in, const int* in_sizes, int n_in,
                              void* d_out, int out_size, void* d_ws, size_t ws_size,
                              hipStream_t stream) {
    const float* x   = (const float*)d_in[0];
    const float* rW  = (const float*)d_in[1];
    const float* rb  = (const float*)d_in[2];
    const float* W1  = (const float*)d_in[3];
    const float* b1  = (const float*)d_in[4];
    const float* W2  = (const float*)d_in[5];
    const float* b2  = (const float*)d_in[6];

    float* out   = (float*)d_out;                       // [B, D_OUT]
    float* probs = out + (size_t)B_TOK * D_OUT;         // [B, E]

    const size_t MB = 1ull << 20;
    uint8_t* ws = (uint8_t*)d_ws;
    unsigned* cnt      = (unsigned*)(ws + 0);
    unsigned* offs     = (unsigned*)(ws + 64);
    unsigned* cursor   = (unsigned*)(ws + 128);
    unsigned* top_idx  = (unsigned*)(ws + 256);                               // B u32
    float2*   top_w    = (float2*)  (ws + 256 + (size_t)B_TOK * 4);           // B float2
    int*      tok_list = (int*)     (ws + 256 + (size_t)B_TOK * 12);          // B*K i32
    float*    gate_lst = (float*)   (ws + 256 + (size_t)B_TOK * 12 + (size_t)B_TOK * TOPK * 4);
    bf16*     xbf      = (bf16*)    (ws + 1 * MB);            // 32 MB
    bf16*     w1bf     = (bf16*)    (ws + 33 * MB);           // 32 MB
    bf16*     w2bf     = (bf16*)    (ws + 65 * MB);           // 32 MB
    bf16*     h_ws     = (bf16*)    (ws + 97 * MB);           // 128 MB

    const int nx4 = B_TOK * D_IN / 4;          // 4,194,304
    const int nw4 = E_NUM * H_DIM * D_IN / 4;  // 4,194,304 (same for W2)
    cvt_bf16_kernel<<<nx4 / 256, 256, 0, stream>>>(x, xbf, nx4);
    cvt_bf16_kernel<<<nw4 / 256, 256, 0, stream>>>(W1, w1bf, nw4);
    cvt_bf16_kernel<<<nw4 / 256, 256, 0, stream>>>(W2, w2bf, nw4);

    moe_zero_kernel<<<1, 32, 0, stream>>>(cnt);
    moe_router_kernel<<<B_TOK / 256, 256, 0, stream>>>(x, rW, rb, probs, top_idx, top_w, cnt);
    moe_prefix_kernel<<<1, 32, 0, stream>>>(cnt, offs, cursor);
    moe_scatter_kernel<<<B_TOK / 256, 256, 0, stream>>>(top_idx, top_w, cursor, tok_list, gate_lst);
    moe_biasinit_kernel<<<B_TOK, 256, 0, stream>>>(b2, top_idx, top_w, out);

    dim3 g1(B_TOK / BM, H_DIM / BN, E_NUM);   // worst-case M tiles; early-exit on count
    moe_gemm1_kernel<<<g1, 256, SMEM_BYTES, stream>>>(xbf, w1bf, b1, tok_list, cnt, offs, h_ws);

    dim3 g2(B_TOK / BM, D_OUT / BN, E_NUM);
    moe_gemm2_kernel<<<g2, 256, SMEM_BYTES, stream>>>(h_ws, w2bf, tok_list, gate_lst, cnt, offs, out);
}
